// MoE_55181739819598
// MI455X (gfx1250) — matrix operand order repaired and checked by reading, not compiled, not run on hardware
//
#include <hip/hip_runtime.h>
#include <hip/hip_bf16.h>
#include <math.h>

typedef __attribute__((ext_vector_type(8)))  float   v8f;
typedef __attribute__((ext_vector_type(8)))  __bf16  v8bf;
typedef __attribute__((ext_vector_type(16))) __bf16  v16bf;

union BF16x16 { v16bf v; v8bf h[2]; };

// Problem constants (B=4, S=1024 -> N=4096 tokens)
constexpr int NTOK = 4096;
constexpr int DIM  = 2048;
constexpr int HID  = 1792;
constexpr int NEXP = 8;

// GEMM tiling: block = 256 threads = 8 waves, wave grid 2(M) x 4(N),
// wave tile 32x32, two WMMA K-slices per LDS stage, double-buffered LDS.
constexpr int BM  = 64;     // tokens per block tile
constexpr int BN  = 128;    // output columns per block tile
constexpr int KC  = 64;     // K-chunk per LDS stage (2 x 32 WMMA slices)
constexpr int LDA = KC + 8; // padded LDS row stride (elements)

// ---------------------------------------------------------------------------
// staging helpers: global -> registers, registers -> (cvt bf16) -> LDS
// ---------------------------------------------------------------------------
__device__ inline void gload16(float4* r, const float* src) {
    const float4* p = (const float4*)src;
#pragma unroll
    for (int i = 0; i < 4; ++i) r[i] = p[i];
}
__device__ inline void gload32(float4* r, const float* src) {
    const float4* p = (const float4*)src;
#pragma unroll
    for (int i = 0; i < 8; ++i) r[i] = p[i];
}
__device__ inline void cvt_store4(__bf16* dst, float4 a, float s) {
    dst[0] = (__bf16)(a.x * s); dst[1] = (__bf16)(a.y * s);
    dst[2] = (__bf16)(a.z * s); dst[3] = (__bf16)(a.w * s);
}
__device__ inline void cvt_store16(__bf16* dst, const float4* r, float s) {
#pragma unroll
    for (int i = 0; i < 4; ++i) cvt_store4(dst + 4 * i, r[i], s);
}
__device__ inline void cvt_store32(__bf16* dst, const float4* r) {
#pragma unroll
    for (int i = 0; i < 8; ++i) cvt_store4(dst + 4 * i, r[i], 1.0f);
}

// ---------------------------------------------------------------------------
// 1) init: zero per-expert counts, zero aux_loss slot of output
// ---------------------------------------------------------------------------
__global__ void moe_init(int* __restrict__ counts, float* __restrict__ out) {
    int t = threadIdx.x;
    if (t < NEXP) counts[t] = 0;
    if (t == NEXP) out[(size_t)NTOK * DIM] = 0.0f;  // aux_loss = 0
}

// ---------------------------------------------------------------------------
// 2) router: sigmoid(x @ gate_w^T), top-2 (+bias for selection), normalize,
//    scatter tokens into per-expert lists. One wave per token (wave32).
// ---------------------------------------------------------------------------
__global__ __launch_bounds__(256)
void moe_router(const float* __restrict__ x,
                const float* __restrict__ gate_w,
                const float* __restrict__ expert_bias,
                int* __restrict__ counts,
                int* __restrict__ token_list,
                float* __restrict__ gate_list) {
    const int lane  = threadIdx.x & 31;
    const int wid   = threadIdx.x >> 5;
    const int token = blockIdx.x * 8 + wid;
    if (token >= NTOK) return;

    const float* xr = x + (size_t)token * DIM;
    float acc[NEXP];
#pragma unroll
    for (int e = 0; e < NEXP; ++e) acc[e] = 0.0f;

    for (int d = lane; d < DIM; d += 32) {
        float xv = xr[d];
#pragma unroll
        for (int e = 0; e < NEXP; ++e)
            acc[e] = fmaf(xv, gate_w[(size_t)e * DIM + d], acc[e]);
    }
#pragma unroll
    for (int e = 0; e < NEXP; ++e) {
#pragma unroll
        for (int off = 16; off > 0; off >>= 1)
            acc[e] += __shfl_xor(acc[e], off, 32);
    }

    if (lane == 0) {
        float s[NEXP];
#pragma unroll
        for (int e = 0; e < NEXP; ++e)
            s[e] = 1.0f / (1.0f + __expf(-acc[e]));
        // top-2 on (score + bias); strict > keeps lowest index on ties
        int i0 = 0; float b0 = -3.0e38f;
#pragma unroll
        for (int e = 0; e < NEXP; ++e) {
            float v = s[e] + expert_bias[e];
            if (v > b0) { b0 = v; i0 = e; }
        }
        int i1 = -1; float b1 = -3.0e38f;
#pragma unroll
        for (int e = 0; e < NEXP; ++e) {
            if (e == i0) continue;
            float v = s[e] + expert_bias[e];
            if (v > b1) { b1 = v; i1 = e; }
        }
        float w0 = s[i0], w1 = s[i1];
        float inv = 1.0f / (w0 + w1 + 1e-20f);
        w0 *= inv; w1 *= inv;

        int slot0 = atomicAdd(&counts[i0], 1);
        token_list[i0 * NTOK + slot0] = token;
        gate_list [i0 * NTOK + slot0] = w0;
        int slot1 = atomicAdd(&counts[i1], 1);
        token_list[i1 * NTOK + slot1] = token;
        gate_list [i1 * NTOK + slot1] = w1;
    }
}

// ---------------------------------------------------------------------------
// 3) exclusive prefix scan of 8 counts -> compacted activation-row offsets
// ---------------------------------------------------------------------------
__global__ void moe_scan(const int* __restrict__ counts, int* __restrict__ offsets) {
    if (threadIdx.x == 0) {
        int run = 0;
        for (int e = 0; e < NEXP; ++e) { offsets[e] = run; run += counts[e]; }
    }
}

// ---------------------------------------------------------------------------
// 4/5) up-projection: act = silu(Xg @ W1^T) * (Xg @ W3^T), bf16 out.
//      expert_mode: rows gathered from per-expert token list, Xg = gate * x.
//      Double-buffered LDS; next chunk's global loads issue under the WMMAs.
// ---------------------------------------------------------------------------
__global__ __launch_bounds__(256)
void moe_up(const float* __restrict__ x,
            const float* __restrict__ w1_base,
            const float* __restrict__ w3_base,
            const int*   __restrict__ counts,
            const int*   __restrict__ offsets,
            const int*   __restrict__ token_list,
            const float* __restrict__ gate_list,
            __bf16* __restrict__ act,
            int expert_mode) {
    const int e    = blockIdx.z;
    const int rows = expert_mode ? counts[e] : NTOK;
    const int m0   = blockIdx.y * BM;
    if (m0 >= rows) return;
    const int n0   = blockIdx.x * BN;

    const float* W1 = w1_base + (expert_mode ? (size_t)e * HID * DIM : 0);
    const float* W3 = w3_base + (expert_mode ? (size_t)e * HID * DIM : 0);

    __shared__ __align__(16) __bf16 As [2][BM][LDA];
    __shared__ __align__(16) __bf16 B1s[2][BN][LDA];
    __shared__ __align__(16) __bf16 B3s[2][BN][LDA];

    const int tid  = threadIdx.x;
    const int lane = tid & 31;
    const int wid  = tid >> 5;
    const int wm   = wid >> 2;   // 0..1  (wave M group, 32 rows each)
    const int wn   = wid & 3;    // 0..3  (wave N group, 32 cols each)

    // A staging: thread -> (row, quarter of KC) -> 16 floats
    const int ra = tid >> 2;     // 0..63
    const int qa = tid & 3;      // 0..3
    const int grow = m0 + ra;
    const bool avalid = grow < rows;
    const float* arow = nullptr;
    float gscale = 0.0f;
    if (avalid) {
        int tk = expert_mode ? token_list[e * NTOK + grow] : grow;
        gscale = expert_mode ? gate_list[e * NTOK + grow] : 1.0f;
        arow   = x + (size_t)tk * DIM + qa * 16;
    } else {
        // zero-fill both buffers once; never rewritten for invalid rows
#pragma unroll
        for (int i = 0; i < 16; ++i) {
            As[0][ra][qa * 16 + i] = (__bf16)0.0f;
            As[1][ra][qa * 16 + i] = (__bf16)0.0f;
        }
    }

    // B staging: thread -> (h row, half of KC) -> 32 floats per matrix
    const int hb    = tid >> 1;  // 0..127
    const int hhalf = tid & 1;
    const float* b1row = W1 + (size_t)(n0 + hb) * DIM + hhalf * 32;
    const float* b3row = W3 + (size_t)(n0 + hb) * DIM + hhalf * 32;

    v8f zero = {0.f, 0.f, 0.f, 0.f, 0.f, 0.f, 0.f, 0.f};
    v8f acc1[2][2], acc3[2][2];   // [A-subtile][N-subtile]
#pragma unroll
    for (int ai = 0; ai < 2; ++ai)
#pragma unroll
        for (int jn = 0; jn < 2; ++jn) { acc1[ai][jn] = zero; acc3[ai][jn] = zero; }

    // per-lane fragment bases (contiguous k = 16*(lane/16) + s, same for A and B)
    const int am0 = wm * 32 + (lane & 15);       // A subtile 0 row
    const int aq  = (lane >> 4) * 16;            // K sub-offset within slice
    const int bn_ = lane & 15;
    const int bq  = (lane >> 4) * 16;

    // register staging for the pipeline
    float4 arg[4], b1g[8], b3g[8];

    // prologue: load + stage chunk 0 into buffer 0
    if (avalid) gload16(arg, arow);
    gload32(b1g, b1row);
    gload32(b3g, b3row);
    if (avalid) cvt_store16(&As[0][ra][qa * 16], arg, gscale);
    cvt_store32(&B1s[0][hb][hhalf * 32], b1g);
    cvt_store32(&B3s[0][hb][hhalf * 32], b3g);
    __syncthreads();

    constexpr int NC = DIM / KC;
    for (int c = 0; c < NC; ++c) {
        const int buf = c & 1;
        const int nk0 = (c + 1) * KC;
        if (c + 1 < NC) {   // issue next chunk's global loads (complete under WMMAs)
            if (avalid) gload16(arg, arow + nk0);
            gload32(b1g, b1row + nk0);
            gload32(b3g, b3row + nk0);
        }

#pragma unroll
        for (int kk = 0; kk < KC; kk += 32) {
            // hoist ALL fragment loads, then burst the 8 WMMAs
            BF16x16 a0, a1;
            a0.h[0] = *(const v8bf*)&As[buf][am0     ][kk + aq];
            a0.h[1] = *(const v8bf*)&As[buf][am0     ][kk + aq + 8];
            a1.h[0] = *(const v8bf*)&As[buf][am0 + 16][kk + aq];
            a1.h[1] = *(const v8bf*)&As[buf][am0 + 16][kk + aq + 8];
            BF16x16 b1f[2], b3f[2];
#pragma unroll
            for (int jn = 0; jn < 2; ++jn) {
                const int col = wn * 32 + jn * 16 + bn_;
                b1f[jn].h[0] = *(const v8bf*)&B1s[buf][col][kk + bq];
                b1f[jn].h[1] = *(const v8bf*)&B1s[buf][col][kk + bq + 8];
                b3f[jn].h[0] = *(const v8bf*)&B3s[buf][col][kk + bq];
                b3f[jn].h[1] = *(const v8bf*)&B3s[buf][col][kk + bq + 8];
            }
#pragma unroll
            for (int jn = 0; jn < 2; ++jn) {
                acc1[0][jn] = __builtin_amdgcn_wmma_f32_16x16x32_bf16(
                    false, a0.v, false, b1f[jn].v, (short)0, acc1[0][jn], false, false);
                acc1[1][jn] = __builtin_amdgcn_wmma_f32_16x16x32_bf16(
                    false, a1.v, false, b1f[jn].v, (short)0, acc1[1][jn], false, false);
                acc3[0][jn] = __builtin_amdgcn_wmma_f32_16x16x32_bf16(
                    false, a0.v, false, b3f[jn].v, (short)0, acc3[0][jn], false, false);
                acc3[1][jn] = __builtin_amdgcn_wmma_f32_16x16x32_bf16(
                    false, a1.v, false, b3f[jn].v, (short)0, acc3[1][jn], false, false);
            }
        }

        if (c + 1 < NC) {   // convert + stage next chunk into the other buffer
            if (avalid) cvt_store16(&As[buf ^ 1][ra][qa * 16], arg, gscale);
            cvt_store32(&B1s[buf ^ 1][hb][hhalf * 32], b1g);
            cvt_store32(&B3s[buf ^ 1][hb][hhalf * 32], b3g);
        }
        __syncthreads();
    }

    // epilogue: act = silu(h1) * h3  (C layout: M = r + 8*(lane/16), N = lane%16)
    const int mhalf = 8 * (lane >> 4);
    const int ncol  = lane & 15;
#pragma unroll
    for (int ai = 0; ai < 2; ++ai) {
#pragma unroll
        for (int jn = 0; jn < 2; ++jn) {
#pragma unroll
            for (int r = 0; r < 8; ++r) {
                const int gm = m0 + wm * 32 + ai * 16 + mhalf + r;
                if (gm < rows) {
                    float h1 = acc1[ai][jn][r];
                    float h3 = acc3[ai][jn][r];
                    float a  = h1 * (1.0f / (1.0f + __expf(-h1))) * h3;
                    const int orow = expert_mode ? (offsets[e] + gm) : gm;
                    act[(size_t)orow * HID + n0 + wn * 32 + jn * 16 + ncol] = (__bf16)a;
                }
            }
        }
    }
}

// ---------------------------------------------------------------------------
// 6/7) down-projection: out(+)= act @ W2^T. shared: plain store; expert:
//      atomicAdd scatter back to owning token. Same pipelined structure.
// ---------------------------------------------------------------------------
__global__ __launch_bounds__(256)
void moe_down(const __bf16* __restrict__ act,
              const float*  __restrict__ w2_base,
              const int*    __restrict__ counts,
              const int*    __restrict__ offsets,
              const int*    __restrict__ token_list,
              float* __restrict__ out,
              int expert_mode) {
    const int e    = blockIdx.z;
    const int rows = expert_mode ? counts[e] : NTOK;
    const int m0   = blockIdx.y * BM;
    if (m0 >= rows) return;
    const int n0   = blockIdx.x * BN;   // over DIM

    const float* W2 = w2_base + (expert_mode ? (size_t)e * DIM * HID : 0);

    __shared__ __align__(16) __bf16 As[2][BM][LDA];
    __shared__ __align__(16) __bf16 Bs[2][BN][LDA];

    const int tid  = threadIdx.x;
    const int lane = tid & 31;
    const int wid  = tid >> 5;
    const int wm   = wid >> 2;   // 0..1
    const int wn   = wid & 3;    // 0..3

    const int ra = tid >> 2;
    const int qa = tid & 3;
    const int grow = m0 + ra;
    const bool avalid = grow < rows;
    const __bf16* arow = nullptr;
    if (avalid) {
        const int orow = expert_mode ? (offsets[e] + grow) : grow;
        arow = act + (size_t)orow * HID + qa * 16;
    } else {
#pragma unroll
        for (int i = 0; i < 16; ++i) {
            As[0][ra][qa * 16 + i] = (__bf16)0.0f;
            As[1][ra][qa * 16 + i] = (__bf16)0.0f;
        }
    }

    const int db    = tid >> 1;
    const int dhalf = tid & 1;
    const float* brow = W2 + (size_t)(n0 + db) * HID + dhalf * 32;

    v8f zero = {0.f, 0.f, 0.f, 0.f, 0.f, 0.f, 0.f, 0.f};
    v8f acc[2][2];
#pragma unroll
    for (int ai = 0; ai < 2; ++ai)
#pragma unroll
        for (int jn = 0; jn < 2; ++jn) acc[ai][jn] = zero;

    const int am0 = wm * 32 + (lane & 15);
    const int aq  = (lane >> 4) * 16;
    const int bn_ = lane & 15;
    const int bq  = (lane >> 4) * 16;

    v8bf ag[2];
    float4 bg[8];

    // prologue: chunk 0 -> buffer 0
    if (avalid) { ag[0] = *(const v8bf*)arow; ag[1] = *(const v8bf*)(arow + 8); }
    gload32(bg, brow);
    if (avalid) {
        *(v8bf*)&As[0][ra][qa * 16]     = ag[0];
        *(v8bf*)&As[0][ra][qa * 16 + 8] = ag[1];
    }
    cvt_store32(&Bs[0][db][dhalf * 32], bg);
    __syncthreads();

    constexpr int NC = HID / KC;
    for (int c = 0; c < NC; ++c) {
        const int buf = c & 1;
        const int nk0 = (c + 1) * KC;
        if (c + 1 < NC) {
            if (avalid) {
                ag[0] = *(const v8bf*)(arow + nk0);
                ag[1] = *(const v8bf*)(arow + nk0 + 8);
            }
            gload32(bg, brow + nk0);
        }

#pragma unroll
        for (int kk = 0; kk < KC; kk += 32) {
            BF16x16 a0, a1;
            a0.h[0] = *(const v8bf*)&As[buf][am0     ][kk + aq];
            a0.h[1] = *(const v8bf*)&As[buf][am0     ][kk + aq + 8];
            a1.h[0] = *(const v8bf*)&As[buf][am0 + 16][kk + aq];
            a1.h[1] = *(const v8bf*)&As[buf][am0 + 16][kk + aq + 8];
            BF16x16 bf[2];
#pragma unroll
            for (int jn = 0; jn < 2; ++jn) {
                const int col = wn * 32 + jn * 16 + bn_;
                bf[jn].h[0] = *(const v8bf*)&Bs[buf][col][kk + bq];
                bf[jn].h[1] = *(const v8bf*)&Bs[buf][col][kk + bq + 8];
            }
#pragma unroll
            for (int jn = 0; jn < 2; ++jn) {
                acc[0][jn] = __builtin_amdgcn_wmma_f32_16x16x32_bf16(
                    false, a0.v, false, bf[jn].v, (short)0, acc[0][jn], false, false);
                acc[1][jn] = __builtin_amdgcn_wmma_f32_16x16x32_bf16(
                    false, a1.v, false, bf[jn].v, (short)0, acc[1][jn], false, false);
            }
        }

        if (c + 1 < NC) {
            if (avalid) {
                *(v8bf*)&As[buf ^ 1][ra][qa * 16]     = ag[0];
                *(v8bf*)&As[buf ^ 1][ra][qa * 16 + 8] = ag[1];
            }
            cvt_store32(&Bs[buf ^ 1][db][dhalf * 32], bg);
        }
        __syncthreads();
    }

    const int mhalf = 8 * (lane >> 4);
    const int ncol  = lane & 15;
#pragma unroll
    for (int ai = 0; ai < 2; ++ai) {
#pragma unroll
        for (int jn = 0; jn < 2; ++jn) {
#pragma unroll
            for (int r = 0; r < 8; ++r) {
                const int gm = m0 + wm * 32 + ai * 16 + mhalf + r;
                if (gm < rows) {
                    const int token = expert_mode ? token_list[e * NTOK + gm] : gm;
                    float* dst = out + (size_t)token * DIM + n0 + wn * 32 + jn * 16 + ncol;
                    if (expert_mode) atomicAdd(dst, acc[ai][jn][r]);
                    else             *dst = acc[ai][jn][r];
                }
            }
        }
    }
}

// ---------------------------------------------------------------------------
// launch
// ---------------------------------------------------------------------------
extern "C" void kernel_launch(void* const* d_in, const int* in_sizes, int n_in,
                              void* d_out, int out_size, void* d_ws, size_t ws_size,
                              hipStream_t stream) {
    (void)in_sizes; (void)n_in; (void)out_size; (void)ws_size;
    const float* x     = (const float*)d_in[0];
    const float* gatew = (const float*)d_in[1];
    const float* w1s   = (const float*)d_in[2];
    const float* w2s   = (const float*)d_in[3];
    const float* w3s   = (const float*)d_in[4];
    const float* w1e   = (const float*)d_in[5];
    const float* w2e   = (const float*)d_in[6];
    const float* w3e   = (const float*)d_in[7];
    const float* ebias = (const float*)d_in[8];
    float* out = (float*)d_out;

    // workspace layout
    int*    counts = (int*)d_ws;                 // [8]
    int*    offs   = counts + NEXP;              // [8]
    int*    tlist  = offs + NEXP;                // [8][4096]
    float*  glist  = (float*)(tlist + NEXP * NTOK);
    __bf16* act_sh = (__bf16*)(glist + NEXP * NTOK);          // [4096][1792] bf16
    __bf16* act_ex = act_sh + (size_t)NTOK * HID;             // [8192][1792] bf16

    moe_init<<<1, 64, 0, stream>>>(counts, out);
    moe_router<<<NTOK / 8, 256, 0, stream>>>(x, gatew, ebias, counts, tlist, glist);
    moe_scan<<<1, 32, 0, stream>>>(counts, offs);

    dim3 gUpS(HID / BN, NTOK / BM, 1);
    moe_up<<<gUpS, 256, 0, stream>>>(x, w1s, w3s, nullptr, nullptr, nullptr, nullptr,
                                     act_sh, 0);
    dim3 gUpE(HID / BN, NTOK / BM, NEXP);
    moe_up<<<gUpE, 256, 0, stream>>>(x, w1e, w3e, counts, offs, tlist, glist,
                                     act_ex, 1);

    dim3 gDnS(DIM / BN, NTOK / BM, 1);
    moe_down<<<gDnS, 256, 0, stream>>>(act_sh, w2s, nullptr, nullptr, nullptr, out, 0);
    dim3 gDnE(DIM / BN, NTOK / BM, NEXP);
    moe_down<<<gDnE, 256, 0, stream>>>(act_ex, w2e, counts, offs, tlist, out, 1);
}
